// f_GRU_54142357733380
// MI455X (gfx1250) — matrix operand order repaired and checked by reading, not compiled, not run on hardware
//
#include <hip/hip_runtime.h>

// ---------------------------------------------------------------------------
// GRU scan for MI455X (gfx1250): persistent grid, bf16 WMMA (f32 accumulate),
// weights pinned in LDS (144KB of the 320KB/WGP), fused input projection,
// grid-wide atomic barrier between timesteps.
// ---------------------------------------------------------------------------

#define T_STEPS 512
#define BATCH   256
#define HID     512
#define IN_DIM  256
#define D_U     128
#define NWG     64
#define WG_THREADS 256
#define TM      64   // batch rows per WG
#define TN      32   // hidden cols per WG

typedef __attribute__((ext_vector_type(16))) __bf16   v16bf;
typedef __attribute__((ext_vector_type(8)))  float    v8f;
typedef __attribute__((ext_vector_type(4)))  float    v4f;
typedef __attribute__((ext_vector_type(4)))  unsigned v4u;

union Frag { v16bf v; v4u q[2]; };

static __device__ __forceinline__ float sigmoid_f(float v) {
    return 1.0f / (1.0f + __expf(-v));
}

static __device__ __forceinline__ v8f wmma_bf16(const Frag& a, const Frag& b, v8f c) {
    // (neg_a, A, neg_b, B, c_mod, C, reuse_a, reuse_b)
    return __builtin_amdgcn_wmma_f32_16x16x32_bf16(false, a.v, false, b.v,
                                                   (short)0, c, false, false);
}

__global__ void gru_init_kernel(const float* __restrict__ x,
                                __bf16* __restrict__ h0,
                                unsigned* __restrict__ bar) {
    int i = blockIdx.x * blockDim.x + threadIdx.x;
    if (i == 0) *bar = 0u;                    // reset grid barrier each launch
    if (i < BATCH * HID) h0[i] = (__bf16)x[i];
}

__global__ void __launch_bounds__(WG_THREADS, 1)
gru_scan_kernel(const float* __restrict__ x,
                const float* __restrict__ U,
                const float* __restrict__ D,
                const float* __restrict__ W_ih,
                const float* __restrict__ W_hh,
                float* __restrict__ out,
                __bf16* __restrict__ hbuf,     // double buffer: 2 * B * H bf16
                unsigned* __restrict__ bar)
{
    extern __shared__ __bf16 smem[];
    __bf16* whh = smem;                        // [3][TN][HID]   (98304 B)
    __bf16* wih = smem + 3 * TN * HID;         // [3][TN][IN_DIM](49152 B)

    const int wg   = blockIdx.x;
    const int mb   = wg >> 4;                  // 0..3  batch block
    const int nb   = wg & 15;                  // 0..15 hidden block
    const int tid  = threadIdx.x;
    const int lane = tid & 31;
    const int wave = tid >> 5;                 // 8 waves
    const int m0   = mb * TM + (wave >> 1) * 16;
    const int nsub = (wave & 1) * 16;
    const int j0   = nb * TN + nsub;

    // ---- pin bf16 weight tiles in LDS (once per launch) ----
    for (int idx = tid; idx < 3 * TN * (HID / 4); idx += WG_THREADS) {
        const int chunk = idx & (HID / 4 - 1);
        const int rc    = idx >> 7;            // g*TN + c
        const int g = rc / TN, c = rc % TN;
        v4f f = *(const v4f*)(W_hh + (size_t)(g * HID + nb * TN + c) * HID + chunk * 4);
        __bf16* p = whh + (size_t)rc * HID + chunk * 4;
        p[0] = (__bf16)f[0]; p[1] = (__bf16)f[1]; p[2] = (__bf16)f[2]; p[3] = (__bf16)f[3];
    }
    for (int idx = tid; idx < 3 * TN * (IN_DIM / 4); idx += WG_THREADS) {
        const int chunk = idx & (IN_DIM / 4 - 1);
        const int rc    = idx >> 6;
        const int g = rc / TN, c = rc % TN;
        v4f f = *(const v4f*)(W_ih + (size_t)(g * HID + nb * TN + c) * IN_DIM + chunk * 4);
        __bf16* p = wih + (size_t)rc * IN_DIM + chunk * 4;
        p[0] = (__bf16)f[0]; p[1] = (__bf16)f[1]; p[2] = (__bf16)f[2]; p[3] = (__bf16)f[3];
    }
    __syncthreads();

    // ISA 16-bit A layout: per-lane K halves at +0 / +16, lanes>=16 shifted by 8.
    // B layout: same as A (per-lane K halves at +0 / +16, lanes>=16 shifted by 8).
    const int arow = lane & 15;
    const int kkA  = (lane >= 16) ? 8 : 0;
    const int kkB  = (lane >= 16) ? 8 : 0;

    const __bf16* bRh = whh + (size_t)(0 * TN + nsub + arow) * HID + kkB;
    const __bf16* bZh = whh + (size_t)(1 * TN + nsub + arow) * HID + kkB;
    const __bf16* bNh = whh + (size_t)(2 * TN + nsub + arow) * HID + kkB;
    const __bf16* bRx = wih + (size_t)(0 * TN + nsub + arow) * IN_DIM + kkB;
    const __bf16* bZx = wih + (size_t)(1 * TN + nsub + arow) * IN_DIM + kkB;
    const __bf16* bNx = wih + (size_t)(2 * TN + nsub + arow) * IN_DIM + kkB;

    for (int t = 0; t < T_STEPS; ++t) {
        const __bf16* hsrc = hbuf + (size_t)(t & 1) * BATCH * HID;
        __bf16*       hdst = hbuf + (size_t)((t + 1) & 1) * BATCH * HID;

        v8f accR = {0, 0, 0, 0, 0, 0, 0, 0};
        v8f accZ = accR, accX = accR, accH = accR;

        // ---- recurrent GEMM: h(t) x W_hh^T  (K = HID, bf16 A from L2) ----
        const __bf16* ah = hsrc + (size_t)(m0 + arow) * HID;
        for (int k = 0; k < HID; k += 32) {
            Frag a, br, bz, bn;
            a.q[0] = *(const v4u*)(ah + k + kkA);
            a.q[1] = *(const v4u*)(ah + k + kkA + 16);
            br.q[0] = *(const v4u*)(bRh + k); br.q[1] = *(const v4u*)(bRh + k + 16);
            bz.q[0] = *(const v4u*)(bZh + k); bz.q[1] = *(const v4u*)(bZh + k + 16);
            bn.q[0] = *(const v4u*)(bNh + k); bn.q[1] = *(const v4u*)(bNh + k + 16);
            accR = wmma_bf16(a, br, accR);
            accZ = wmma_bf16(a, bz, accZ);
            accH = wmma_bf16(a, bn, accH);
        }

        // ---- fused input GEMM: [U|D](t) x W_ih^T (fp32 A converted inline) ----
        const size_t rowUD = ((size_t)t * BATCH + (m0 + arow)) * D_U;
        if (t + 1 < T_STEPS) {
            __builtin_prefetch(U + rowUD + (size_t)BATCH * D_U, 0, 1);
            __builtin_prefetch(D + rowUD + (size_t)BATCH * D_U, 0, 1);
        }
        for (int k = 0; k < IN_DIM; k += 32) {
            Frag a, br, bz, bn;
            const int k0 = k + kkA;          // 8-aligned, never straddles U|D seam
            const int k1 = k0 + 16;
            const float* p0 = (k0 < D_U) ? (U + rowUD + k0) : (D + rowUD + (k0 - D_U));
            const float* p1 = (k1 < D_U) ? (U + rowUD + k1) : (D + rowUD + (k1 - D_U));
            v4f f0 = *(const v4f*)p0, f1 = *(const v4f*)(p0 + 4);
            v4f f2 = *(const v4f*)p1, f3 = *(const v4f*)(p1 + 4);
            #pragma unroll
            for (int i = 0; i < 4; ++i) {
                a.v[i]      = (__bf16)f0[i];
                a.v[4 + i]  = (__bf16)f1[i];
                a.v[8 + i]  = (__bf16)f2[i];
                a.v[12 + i] = (__bf16)f3[i];
            }
            br.q[0] = *(const v4u*)(bRx + k); br.q[1] = *(const v4u*)(bRx + k + 16);
            bz.q[0] = *(const v4u*)(bZx + k); bz.q[1] = *(const v4u*)(bZx + k + 16);
            bn.q[0] = *(const v4u*)(bNx + k); bn.q[1] = *(const v4u*)(bNx + k + 16);
            accR = wmma_bf16(a, br, accR);
            accZ = wmma_bf16(a, bz, accZ);
            accX = wmma_bf16(a, bn, accX);
        }

        // ---- gates + state update (C/D layout: lane%16 = N, vgpr e + 8*(lane>=16) = M) ----
        const int j     = j0 + arow;
        const int mbase = m0 + ((lane >= 16) ? 8 : 0);
        #pragma unroll
        for (int e = 0; e < 8; ++e) {
            const int b = mbase + e;
            const float hprev = (t == 0)
                ? x[(size_t)b * HID + j]
                : out[((size_t)(t - 1) * BATCH + b) * HID + j];
            const float r  = sigmoid_f(accR[e]);
            const float z  = sigmoid_f(accZ[e]);
            const float n  = tanhf(accX[e] + r * accH[e]);
            const float hn = (1.0f - z) * n + z * hprev;
            const size_t o = ((size_t)t * BATCH + b) * HID + j;
            out[o] = hn;                                        // X
            out[(size_t)T_STEPS * BATCH * HID + o] = hn;        // Y duplicate
            hdst[(size_t)b * HID + j] = (__bf16)hn;             // next-step A matrix
        }

        // ---- grid-wide barrier: h(t+1) visible before anyone reads it ----
        if (t + 1 < T_STEPS) {
            __threadfence();
            __syncthreads();
            if (tid == 0) {
                __hip_atomic_fetch_add(bar, 1u, __ATOMIC_ACQ_REL, __HIP_MEMORY_SCOPE_AGENT);
                const unsigned target = (unsigned)(t + 1) * NWG;
                while (__hip_atomic_load(bar, __ATOMIC_ACQUIRE, __HIP_MEMORY_SCOPE_AGENT) < target) {
                    __builtin_amdgcn_s_sleep(2);
                }
            }
            __syncthreads();
            __threadfence();
        }
    }
}

extern "C" void kernel_launch(void* const* d_in, const int* in_sizes, int n_in,
                              void* d_out, int out_size, void* d_ws, size_t ws_size,
                              hipStream_t stream) {
    (void)in_sizes; (void)n_in; (void)out_size; (void)ws_size;
    const float* x    = (const float*)d_in[0];
    const float* U    = (const float*)d_in[1];
    const float* D    = (const float*)d_in[2];
    const float* W_ih = (const float*)d_in[3];
    const float* W_hh = (const float*)d_in[4];
    float* out = (float*)d_out;

    unsigned* bar = (unsigned*)d_ws;                       // barrier counter
    __bf16* hbuf  = (__bf16*)((char*)d_ws + 256);          // 2*B*H bf16 double buffer

    const size_t smem_bytes = (size_t)(3 * TN * HID + 3 * TN * IN_DIM) * sizeof(__bf16); // 147456
    (void)hipFuncSetAttribute(reinterpret_cast<const void*>(gru_scan_kernel),
                              hipFuncAttributeMaxDynamicSharedMemorySize,
                              (int)smem_bytes);

    gru_init_kernel<<<(BATCH * HID + 255) / 256, 256, 0, stream>>>(x, hbuf, bar);
    gru_scan_kernel<<<NWG, WG_THREADS, smem_bytes, stream>>>(x, U, D, W_ih, W_hh,
                                                             out, hbuf, bar);
}
